// CrossAttention_24438363914471
// MI455X (gfx1250) — matrix operand order repaired and checked by reading, not compiled, not run on hardware
//
#include <hip/hip_runtime.h>
#include <hip/hip_bf16.h>
#include <cstdint>
#include <cstddef>

namespace {

constexpr int kB = 8;
constexpr int kC = 512;
constexpr int kN = 48 * 48;        // 2304 tokens
constexpr int kQT = kN / 16;       // 144 token tiles
constexpr int kCT = kC / 16;       // 32 channel tiles

typedef __attribute__((ext_vector_type(16))) __bf16 bf16x16;
typedef __attribute__((ext_vector_type(8)))  __bf16 bf16x8;
typedef __attribute__((ext_vector_type(8)))  float  f32x8;

__device__ __forceinline__ f32x8 wmma_bf16(bf16x16 a, bf16x16 b, f32x8 c) {
  return __builtin_amdgcn_wmma_f32_16x16x32_bf16(false, a, false, b, (short)0, c,
                                                 false, false);
}

// A fragment (16x32 bf16): lane L -> row m = L&15; element j -> K=hi*16+j.
// Rows are contiguous (rowStride elements apart).
__device__ __forceinline__ bf16x16 loadA(const __bf16* base, size_t rowStride,
                                         int ln, int hi, int ci) {
  const __bf16* p = base + (size_t)ln * rowStride + ci + hi * 16;
  bf16x8 lo = *(const bf16x8*)p;
  bf16x8 hp = *(const bf16x8*)(p + 8);
  bf16x16 a;
#pragma unroll
  for (int j = 0; j < 8; ++j) { a[j] = lo[j]; a[8 + j] = hp[j]; }
  return a;
}

// ---------------------------------------------------------------- weights ----
__global__ __launch_bounds__(256) void cvt_w_kernel(
    const float* __restrict__ Wq, const float* __restrict__ Wk,
    const float* __restrict__ Wv, __bf16* __restrict__ Wqb,
    __bf16* __restrict__ Wkb, __bf16* __restrict__ Wvb) {
  int i = blockIdx.x * 256 + threadIdx.x;
  Wqb[i] = (__bf16)Wq[i];
  Wkb[i] = (__bf16)Wk[i];
  Wvb[i] = (__bf16)Wv[i];
}

// wEff[c'] = sum_c Wconv[c] * Wo[c][c'];  wEff[kC] = Wconv.bo + bconv
__global__ __launch_bounds__(256) void weff_kernel(
    const float* __restrict__ Wo, const float* __restrict__ bo,
    const float* __restrict__ Wconv, const float* __restrict__ bconv,
    float* __restrict__ wEff) {
  int cp = blockIdx.x * 256 + threadIdx.x;
  if (cp < kC) {
    float s = 0.f;
    for (int c = 0; c < kC; ++c) s += Wconv[c] * Wo[(size_t)c * kC + cp];
    wEff[cp] = s;
  }
  if (cp == 0) {
    float s = bconv[0];
    for (int c = 0; c < kC; ++c) s += Wconv[c] * bo[c];
    wEff[kC] = s;
  }
}

// ------------------------------------------- transpose [B,C,N]f32 -> [B,N,C]bf16
__global__ __launch_bounds__(256) void transpose_cvt_kernel(
    const float* __restrict__ in, __bf16* __restrict__ out) {
  __shared__ float t[32][65];
  int b = blockIdx.z;
  int c0 = blockIdx.y * 32;
  int n0 = blockIdx.x * 64;
  for (int i = threadIdx.x; i < 32 * 64; i += 256) {
    int ci = i >> 6, ni = i & 63;
    t[ci][ni] = in[((size_t)b * kC + c0 + ci) * kN + n0 + ni];
  }
  __syncthreads();
  for (int i = threadIdx.x; i < 32 * 64; i += 256) {
    int ci = i & 31, ni = i >> 5;
    out[((size_t)b * kN + n0 + ni) * kC + c0 + ci] = (__bf16)t[ci][ni];
  }
}

// ------------------------- d_out base: Wconv . lf[b,:,n]  + bEff  (coalesced n)
__global__ __launch_bounds__(256) void base_kernel(
    const float* __restrict__ lf, const float* __restrict__ Wconv,
    const float* __restrict__ wEff, float* __restrict__ dout) {
  int b = blockIdx.y;
  int n = blockIdx.x * 256 + threadIdx.x;
  float s = wEff[kC];
  const float* p = lf + (size_t)b * kC * kN + n;
  for (int c = 0; c < kC; ++c) s += Wconv[c] * p[(size_t)c * kN];
  dout[(size_t)b * kN + n] = s;
}

// ---------------- Q/K projection: Out[b][n][co] = X[b][n][:] . W[co][:] + bias
__global__ __launch_bounds__(256) void proj_nc_kernel(
    const __bf16* __restrict__ X, const __bf16* __restrict__ Wb,
    const float* __restrict__ bias, __bf16* __restrict__ Out) {
  const int w = threadIdx.x >> 5;
  const int lane = threadIdx.x & 31;
  const int ln = lane & 15, hi = lane >> 4;
  int tile = blockIdx.x * 8 + w;
  int ct = tile & (kCT - 1);
  int nt = (tile / kCT) % kQT;
  int b  = tile / (kCT * kQT);
  const __bf16* Arow = X + ((size_t)b * kN + nt * 16) * kC;
  const __bf16* Bcol = Wb + (size_t)(ct * 16 + ln) * kC + hi * 16;
  const f32x8 zero = {0, 0, 0, 0, 0, 0, 0, 0};
  f32x8 acc = zero;
#pragma unroll 4
  for (int ci = 0; ci < kC; ci += 32) {
    bf16x16 a = loadA(Arow, kC, ln, hi, ci);
    bf16x16 bb = *(const bf16x16*)(Bcol + ci);
    acc = wmma_bf16(a, bb, acc);
  }
  float bvv = bias[ct * 16 + ln];
  __bf16* orow = Out + ((size_t)b * kN + nt * 16 + hi * 8) * kC + ct * 16 + ln;
#pragma unroll
  for (int r = 0; r < 8; ++r) orow[(size_t)r * kC] = (__bf16)(acc[r] + bvv);
}

// --- V projection stored transposed: Vt[b][co][n] = G[b][n][:] . Wv[co][:] + bv
__global__ __launch_bounds__(256) void proj_v_kernel(
    const __bf16* __restrict__ G, const __bf16* __restrict__ Wvb,
    const float* __restrict__ bv, __bf16* __restrict__ VtOut) {
  const int w = threadIdx.x >> 5;
  const int lane = threadIdx.x & 31;
  const int ln = lane & 15, hi = lane >> 4;
  int tile = blockIdx.x * 8 + w;
  int nt = tile % kQT;
  int ct = (tile / kQT) % kCT;
  int b  = tile / (kQT * kCT);
  const __bf16* Arow = Wvb + (size_t)(ct * 16) * kC;                    // rows = co
  const __bf16* Brow = G + ((size_t)b * kN + nt * 16 + ln) * kC + hi * 16;
  const f32x8 zero = {0, 0, 0, 0, 0, 0, 0, 0};
  f32x8 acc = zero;
#pragma unroll 4
  for (int ci = 0; ci < kC; ci += 32) {
    bf16x16 a = loadA(Arow, kC, ln, hi, ci);
    bf16x16 bb = *(const bf16x16*)(Brow + ci);
    acc = wmma_bf16(a, bb, acc);
  }
#pragma unroll
  for (int r = 0; r < 8; ++r) {
    int co = ct * 16 + hi * 8 + r;
    VtOut[((size_t)b * kC + co) * kN + nt * 16 + ln] = (__bf16)(acc[r] + bv[co]);
  }
}

// ------------------------------------------------ fused flash attention ------
// 1 workgroup = 4 waves handles (b, 16-q tile). Per 128-key iteration:
//   each wave: S for its 32 keys (32 WMMAs) -> cross-wave softmax via LDS ->
//   P(bf16) to LDS -> each wave: P.V for its 128-channel slice (32 WMMAs).
// Epilogue: O row dot wEff / rowsum, deterministic cross-wave sum into d_out.
__global__ __launch_bounds__(128) void attn_kernel(
    const __bf16* __restrict__ Qb, const __bf16* __restrict__ Kb,
    const __bf16* __restrict__ Vt, const float* __restrict__ wEff,
    float* __restrict__ dout) {
  const int b = blockIdx.y;
  const int q0 = blockIdx.x * 16;
  const int w = threadIdx.x >> 5;
  const int lane = threadIdx.x & 31;
  const int ln = lane & 15;
  const int hi = lane >> 4;

  __shared__ __bf16 Qs[16][kC];       // 16 KB
  __shared__ __bf16 Pbuf[16][128];    // 4 KB
  __shared__ float maxbuf[4][16];
  __shared__ float sumbuf[4][16];
  __shared__ float outbuf[4][16];

  const __bf16* Qg = Qb + ((size_t)b * kN + q0) * kC;
  for (int i = threadIdx.x; i < 16 * kC / 8; i += 128) {
    int row = i >> 6;
    int col = (i & 63) << 3;
    *(bf16x8*)&Qs[row][col] = *(const bf16x8*)(Qg + (size_t)row * kC + col);
  }
  __syncthreads();

  const f32x8 zero = {0, 0, 0, 0, 0, 0, 0, 0};
  f32x8 acc[8];
#pragma unroll
  for (int t = 0; t < 8; ++t) acc[t] = zero;
  float Mrun[8], Lrun[8];
#pragma unroll
  for (int r = 0; r < 8; ++r) { Mrun[r] = -1.0e30f; Lrun[r] = 0.f; }

  const __bf16* Vw = Vt + ((size_t)b * kC + w * 128) * kN;

  for (int kb = 0; kb < kN; kb += 128) {
    // S = Q K^T for this wave's 32 keys (two 16x16 tiles)
    f32x8 S0 = zero, S1 = zero;
    const __bf16* Krow = Kb + ((size_t)b * kN + kb + w * 32) * kC;
#pragma unroll 4
    for (int ci = 0; ci < kC; ci += 32) {
      bf16x16 a = loadA(&Qs[0][0], kC, ln, hi, ci);
      bf16x16 b0 = *(const bf16x16*)(Krow + (size_t)ln * kC + ci + hi * 16);
      bf16x16 b1 = *(const bf16x16*)(Krow + (size_t)(16 + ln) * kC + ci + hi * 16);
      S0 = wmma_bf16(a, b0, S0);
      S1 = wmma_bf16(a, b1, S1);
    }
    // per-wave row max over 32 keys (row m = 8*hi + r lives in lane half hi)
    float mv[8];
#pragma unroll
    for (int r = 0; r < 8; ++r) mv[r] = fmaxf(S0[r], S1[r]);
#pragma unroll
    for (int d = 1; d < 16; d <<= 1)
#pragma unroll
      for (int r = 0; r < 8; ++r) mv[r] = fmaxf(mv[r], __shfl_xor(mv[r], d, 16));
    if (ln == 0) {
#pragma unroll
      for (int r = 0; r < 8; ++r) maxbuf[w][hi * 8 + r] = mv[r];
    }
    __syncthreads();
    float mnew[8], scale[8];
#pragma unroll
    for (int r = 0; r < 8; ++r) {
      int row = hi * 8 + r;
      float bm = fmaxf(fmaxf(maxbuf[0][row], maxbuf[1][row]),
                       fmaxf(maxbuf[2][row], maxbuf[3][row]));
      mnew[r] = fmaxf(Mrun[r], bm);
      scale[r] = __expf(Mrun[r] - mnew[r]);
      Mrun[r] = mnew[r];
    }
    // P = exp(S - m); stash bf16 P in LDS; partial row sums
    float ps[8];
#pragma unroll
    for (int r = 0; r < 8; ++r) {
      float p0 = __expf(S0[r] - mnew[r]);
      float p1 = __expf(S1[r] - mnew[r]);
      ps[r] = p0 + p1;
      Pbuf[hi * 8 + r][w * 32 + ln] = (__bf16)p0;
      Pbuf[hi * 8 + r][w * 32 + 16 + ln] = (__bf16)p1;
    }
#pragma unroll
    for (int d = 1; d < 16; d <<= 1)
#pragma unroll
      for (int r = 0; r < 8; ++r) ps[r] += __shfl_xor(ps[r], d, 16);
    if (ln == 0) {
#pragma unroll
      for (int r = 0; r < 8; ++r) sumbuf[w][hi * 8 + r] = ps[r];
    }
    __syncthreads();
#pragma unroll
    for (int r = 0; r < 8; ++r) {
      int row = hi * 8 + r;
      Lrun[r] = Lrun[r] * scale[r] +
                (sumbuf[0][row] + sumbuf[1][row] + sumbuf[2][row] + sumbuf[3][row]);
    }
    // rescale accumulators, then O += P.V for this wave's 128 channels
#pragma unroll
    for (int t = 0; t < 8; ++t)
#pragma unroll
      for (int r = 0; r < 8; ++r) acc[t][r] *= scale[r];
#pragma unroll
    for (int kt = 0; kt < 4; ++kt) {
      bf16x16 a = loadA(&Pbuf[0][0], 128, ln, hi, kt * 32);
#pragma unroll
      for (int ct = 0; ct < 8; ++ct) {
        bf16x16 bb = *(const bf16x16*)(Vw + (size_t)(ct * 16 + ln) * kN + kb +
                                       kt * 32 + hi * 16);
        acc[ct] = wmma_bf16(a, bb, acc[ct]);
      }
    }
    __syncthreads();   // Pbuf/maxbuf reused next iteration
  }

  // epilogue: rowdot(wEff, O)/L, deterministic cross-wave reduction into d_out
  float rd[8];
#pragma unroll
  for (int r = 0; r < 8; ++r) rd[r] = 0.f;
#pragma unroll
  for (int ct = 0; ct < 8; ++ct) {
    float we = wEff[w * 128 + ct * 16 + ln];
#pragma unroll
    for (int r = 0; r < 8; ++r) rd[r] += we * acc[ct][r];
  }
#pragma unroll
  for (int d = 1; d < 16; d <<= 1)
#pragma unroll
    for (int r = 0; r < 8; ++r) rd[r] += __shfl_xor(rd[r], d, 16);
  if (ln == 0) {
#pragma unroll
    for (int r = 0; r < 8; ++r) outbuf[w][hi * 8 + r] = rd[r] / Lrun[r];
  }
  __syncthreads();
  if (threadIdx.x < 16) {
    float v = outbuf[0][threadIdx.x] + outbuf[1][threadIdx.x] +
              outbuf[2][threadIdx.x] + outbuf[3][threadIdx.x];
    size_t oi = (size_t)b * kN + q0 + threadIdx.x;
    dout[oi] += v;   // base written by base_kernel earlier on the stream
  }
}

}  // namespace

extern "C" void kernel_launch(void* const* d_in, const int* in_sizes, int n_in,
                              void* d_out, int out_size, void* d_ws, size_t ws_size,
                              hipStream_t stream) {
  (void)in_sizes; (void)n_in; (void)out_size; (void)ws_size;
  const float* lf    = (const float*)d_in[0];
  const float* gf    = (const float*)d_in[1];
  const float* Wq    = (const float*)d_in[2];
  const float* bq    = (const float*)d_in[3];
  const float* Wk    = (const float*)d_in[4];
  const float* bk    = (const float*)d_in[5];
  const float* Wv    = (const float*)d_in[6];
  const float* bv    = (const float*)d_in[7];
  const float* Wo    = (const float*)d_in[8];
  const float* bo    = (const float*)d_in[9];
  const float* Wconv = (const float*)d_in[10];
  const float* bconv = (const float*)d_in[11];
  float* out = (float*)d_out;

  char* ws = (char*)d_ws;
  size_t off = 0;
  auto take = [&](size_t bytes) -> char* {
    char* p = ws + off;
    off += (bytes + 255) & ~(size_t)255;
    return p;
  };
  const size_t actBytes = (size_t)kB * kN * kC * sizeof(__bf16);   // 18.87 MB
  __bf16* Xt   = (__bf16*)take(actBytes);
  __bf16* Gt   = (__bf16*)take(actBytes);
  __bf16* Qb   = (__bf16*)take(actBytes);
  __bf16* Kbuf = (__bf16*)take(actBytes);
  __bf16* Vt   = (__bf16*)take(actBytes);
  __bf16* Wqb  = (__bf16*)take((size_t)kC * kC * sizeof(__bf16));
  __bf16* Wkb  = (__bf16*)take((size_t)kC * kC * sizeof(__bf16));
  __bf16* Wvb  = (__bf16*)take((size_t)kC * kC * sizeof(__bf16));
  float*  wEff = (float*)take((size_t)(kC + 1) * sizeof(float));

  cvt_w_kernel<<<kC * kC / 256, 256, 0, stream>>>(Wq, Wk, Wv, Wqb, Wkb, Wvb);
  weff_kernel<<<2, 256, 0, stream>>>(Wo, bo, Wconv, bconv, wEff);
  dim3 tg(kN / 64, kC / 32, kB);
  transpose_cvt_kernel<<<tg, 256, 0, stream>>>(lf, Xt);
  transpose_cvt_kernel<<<tg, 256, 0, stream>>>(gf, Gt);
  base_kernel<<<dim3(kN / 256, kB), 256, 0, stream>>>(lf, Wconv, wEff, out);

  int projBlocks = kB * kQT * kCT / 8;   // 4608 blocks x 8 waves
  proj_nc_kernel<<<projBlocks, 256, 0, stream>>>(Xt, Wqb, bq, Qb);
  proj_nc_kernel<<<projBlocks, 256, 0, stream>>>(Gt, Wkb, bk, Kbuf);
  proj_v_kernel<<<projBlocks, 256, 0, stream>>>(Gt, Wvb, bv, Vt);

  attn_kernel<<<dim3(kQT, kB), 128, 0, stream>>>(Qb, Kbuf, Vt, wEff, out);
}
